// NaiveConv_10746008175003
// MI455X (gfx1250) — compile-verified, matrix operand order checked by reading, not run on hardware
//
#include <hip/hip_runtime.h>
#include <hip/hip_bf16.h>

// out = T @ x,  T[i,j] = a[i-j] (i>=j), n=8192, d=1024, f32 in/out.
// Prep kernel: transpose+convert x -> xT_bf[d][n] bf16 in d_ws.
// Main kernel: bf16 WMMA GEMM; Toeplitz A fragments from reversed zero-padded
// bf16 copy of `a` in LDS (pad == causal mask); B tiles DMA'd into LDS by the
// Tensor Data Mover (double buffered, TENSORcnt-synchronized).

typedef __attribute__((ext_vector_type(16))) __bf16 v16bf;
typedef __attribute__((ext_vector_type(8)))  float  v8f;
typedef __attribute__((ext_vector_type(4)))  unsigned int v4u;
typedef __attribute__((ext_vector_type(8)))  int  v8i;
typedef __attribute__((ext_vector_type(4)))  int  v4i;

union ABFrag { v16bf v; unsigned int u[8]; };

__device__ __forceinline__ unsigned short f32_to_bf16_bits(float f) {
  __bf16 h = (__bf16)f;                       // RNE fptrunc -> v_cvt
  return __builtin_bit_cast(unsigned short, h);
}

#define MT 128      // workgroup M tile
#define NT 128      // workgroup N tile
#define KT 32       // K step (bf16 WMMA K)
#define N_A 8192    // kernel length (== n)
#define A_PAD 160   // zero pad: j_max = 8191 + 96 + 31 = 8318 < 8192+160

#if defined(__gfx1250__) && __has_builtin(__builtin_amdgcn_tensor_load_to_lds) && \
    __has_builtin(__builtin_amdgcn_s_wait_tensorcnt)
#define HAVE_TDM 1
#else
#define HAVE_TDM 0
#endif

#if HAVE_TDM
// One 2D TDM descriptor: tile KT(k, contiguous) x NT(n) of bf16 from
// xbf[d][n] (row stride n elements) into LDS at lds_off, [n][k] contiguous.
__device__ __forceinline__ void tdm_load_tile(const unsigned short* xbf,
                                              int n, int d, int n0, int k0,
                                              unsigned lds_off) {
  unsigned long long ga = (unsigned long long)(size_t)xbf +
      ((unsigned long long)(unsigned)n0 * (unsigned)n + (unsigned)k0) * 2ull;
  v4u g0;
  g0.x = 1u;                                   // count=1, user mode
  g0.y = lds_off;                              // LDS byte address
  g0.z = (unsigned)ga;                         // global_addr[31:0]
  g0.w = (unsigned)(ga >> 32) | 0x80000000u;   // addr[56:32] | type=2<<30
  v8i g1;
  g1[0] = 0x00010000;                          // wg_mask=0, data_size=1 (2B)
  g1[1] = (int)(((unsigned)n & 0xFFFFu) << 16);        // tensor_dim0 lo16
  g1[2] = (int)(((unsigned)n >> 16) |
                (((unsigned)d & 0xFFFFu) << 16));      // dim0 hi16 | dim1 lo16
  g1[3] = (int)(((unsigned)d >> 16) | ((unsigned)KT << 16)); // dim1 hi | tile0
  g1[4] = NT;                                  // tile_dim1 (tile_dim2=0)
  g1[5] = n;                                   // tensor_dim0_stride[31:0]
  g1[6] = 0;                                   // stride hi | dim1_stride lo
  g1[7] = 0;
  v4i z4 = {0, 0, 0, 0};
  v8i z8 = {0, 0, 0, 0, 0, 0, 0, 0};
  __builtin_amdgcn_tensor_load_to_lds(g0, g1, z4, z4, z8, 0);
}
#endif

// ---------------- prep: x[n][d] f32 -> xbf[d][n] bf16 (LDS-tiled transpose)
__global__ __launch_bounds__(256, 1) void transpose_cvt_bf16(
    const float* __restrict__ x, unsigned short* __restrict__ xbf,
    int n, int d) {
  __shared__ unsigned short t[32][33];
  const int tx = threadIdx.x & 31;
  const int ty = threadIdx.x >> 5;            // 0..7
  const int c0 = blockIdx.x * 32;             // col base (d dim)
  const int r0 = blockIdx.y * 32;             // row base (n dim)
#pragma unroll
  for (int j = 0; j < 4; ++j) {
    const int r = ty + j * 8;
    t[tx][r] = f32_to_bf16_bits(x[(size_t)(r0 + r) * d + c0 + tx]);
  }
  __syncthreads();
#pragma unroll
  for (int j = 0; j < 4; ++j) {
    const int c = ty + j * 8;
    xbf[(size_t)(c0 + c) * n + r0 + tx] = t[c][tx];
  }
}

// ---------------- main GEMM ------------------------------------------------
template <bool USE_TDM>
__global__ __launch_bounds__(256, 1) void toeplitz_bf16_wmma(
    const float* __restrict__ a, const float* __restrict__ x,
    const unsigned short* __restrict__ xbf, float* __restrict__ out,
    int n, int d) {
  __shared__ __align__(16) unsigned short arev[N_A + A_PAD];
  __shared__ __align__(16) unsigned short xT[2][NT * KT];  // [n][k] bf16

  const int tid = threadIdx.x;
  const int m0 = blockIdx.y * MT;
  const int n0 = blockIdx.x * NT;
  const int wave = tid >> 5;
  const int lane = tid & 31;
  const int wm   = (wave & 3) * 32;   // wave M offset (4 waves in M)
  const int wn   = (wave >> 2) * 64;  // wave N offset (2 waves in N)
  const int lm   = lane & 15;
  const int half = lane >> 4;
  const int kb   = half * 8;          // K base per 16-bit operand layout

  const int kend = m0 + MT;           // causal: only k0 <= m0+127 contributes
  int cur = 0;

#if HAVE_TDM
  if constexpr (USE_TDM) {
    if (wave == 0)
      tdm_load_tile(xbf, n, d, n0, 0, (unsigned)(size_t)&xT[0][0]);
  }
#endif

  // Stage reversed bf16 kernel + zero pad (pad = causal mask); overlaps TDM.
  for (int j = tid; j < N_A; j += 256)
    arev[j] = f32_to_bf16_bits(a[N_A - 1 - j]);
  for (int j = N_A + tid; j < N_A + A_PAD; j += 256)
    arev[j] = 0;

#if HAVE_TDM
  if constexpr (USE_TDM) {
    if (wave == 0) __builtin_amdgcn_s_wait_tensorcnt(0);
  }
#endif
  __syncthreads();

  v8f acc[2][4] = {};

  for (int k0 = 0; k0 < kend; k0 += KT) {
    if constexpr (USE_TDM) {
#if HAVE_TDM
      if (k0 + KT < kend && wave == 0)
        tdm_load_tile(xbf, n, d, n0, k0 + KT,
                      (unsigned)(size_t)&xT[cur ^ 1][0]);
#endif
    } else {
      // Staging fallback: global f32 -> cvt -> LDS bf16 (single buffer).
#pragma unroll
      for (int it = 0; it < 4; ++it) {
        const int idx = tid + it * 256;       // 1024 float4 groups
        const int kk  = idx >> 5;             // 0..31
        const int gg  = (idx & 31) << 2;      // 0,4,...,124
        const float4 f4 =
            *(const float4*)(x + (size_t)(k0 + kk) * d + n0 + gg);
        xT[0][(gg + 0) * KT + kk] = f32_to_bf16_bits(f4.x);
        xT[0][(gg + 1) * KT + kk] = f32_to_bf16_bits(f4.y);
        xT[0][(gg + 2) * KT + kk] = f32_to_bf16_bits(f4.z);
        xT[0][(gg + 3) * KT + kk] = f32_to_bf16_bits(f4.w);
      }
      __syncthreads();
    }

    const unsigned short* xcur = &xT[USE_TDM ? cur : 0][0];

    // A fragments: 16x32 bf16 per ISA layout; VGPR v holds K pair {K2,K2+1};
    // K2 = kb+2v (v<4) or 16+kb+2(v-4).  A[m][K] = arev[(n-1)-m_g+k_g].
    ABFrag afr[2];
#pragma unroll
    for (int sm = 0; sm < 2; ++sm) {
      const int mg = m0 + wm + sm * 16 + lm;
      const unsigned short* p = arev + (N_A - 1 - mg + k0 + kb);
#pragma unroll
      for (int v = 0; v < 8; ++v) {
        const int off = (v < 4) ? (v << 1) : (16 + ((v - 4) << 1));
        afr[sm].u[v] = (unsigned)p[off] | ((unsigned)p[off + 1] << 16);
      }
    }

    // B fragments: 32x16 bf16, column per lane, K striped like A.
    ABFrag bfr[4];
#pragma unroll
    for (int sn = 0; sn < 4; ++sn) {
      const unsigned short* q = xcur + (wn + sn * 16 + lm) * KT + kb;
#pragma unroll
      for (int v = 0; v < 8; ++v) {
        const int off = (v < 4) ? (v << 1) : (16 + ((v - 4) << 1));
        bfr[sn].u[v] = *(const unsigned int*)(q + off);  // 4B aligned
      }
    }

#pragma unroll
    for (int sm = 0; sm < 2; ++sm)
#pragma unroll
      for (int sn = 0; sn < 4; ++sn)
        acc[sm][sn] = __builtin_amdgcn_wmma_f32_16x16x32_bf16(
            false, afr[sm].v, false, bfr[sn].v,
            (short)0, acc[sm][sn], false, false);

    if constexpr (USE_TDM) {
#if HAVE_TDM
      if (wave == 0) __builtin_amdgcn_s_wait_tensorcnt(0);
#endif
      __syncthreads();     // next tile visible; reads of cur complete
      cur ^= 1;
    } else {
      __syncthreads();
    }
  }

  // Epilogue: C/D layout -> row = base + r + 8*half, col = base + lm.
#pragma unroll
  for (int sm = 0; sm < 2; ++sm) {
    const int row0 = m0 + wm + sm * 16 + half * 8;
#pragma unroll
    for (int sn = 0; sn < 4; ++sn) {
      const int col = n0 + wn + sn * 16 + lm;
#pragma unroll
      for (int r = 0; r < 8; ++r)
        out[(size_t)(row0 + r) * d + col] = acc[sm][sn][r];
    }
  }
}

extern "C" void kernel_launch(void* const* d_in, const int* in_sizes, int n_in,
                              void* d_out, int out_size, void* d_ws, size_t ws_size,
                              hipStream_t stream) {
  const float* a = (const float*)d_in[0];
  const float* x = (const float*)d_in[1];
  float* out = (float*)d_out;
  const int n = in_sizes[0];          // 8192
  const int d = in_sizes[1] / n;      // 1024

  const size_t need = (size_t)n * (size_t)d * sizeof(unsigned short);
  unsigned short* xbf = (unsigned short*)d_ws;
  const bool use_tdm = (ws_size >= need);

  dim3 ggemm(d / NT, n / MT);         // (8, 64)
  if (use_tdm) {
    dim3 gtr(d / 32, n / 32);         // (32, 256)
    transpose_cvt_bf16<<<gtr, 256, 0, stream>>>(x, xbf, n, d);
    toeplitz_bf16_wmma<true><<<ggemm, 256, 0, stream>>>(a, x, xbf, out, n, d);
  } else {
    toeplitz_bf16_wmma<false><<<ggemm, 256, 0, stream>>>(a, x, xbf, out, n, d);
  }
}
